// CoordNet_71390946394556
// MI455X (gfx1250) — compile-verified, matrix operand order checked by reading, not run on hardware
//
#include <hip/hip_runtime.h>
#include <hip/hip_bf16.h>
#include <math.h>

// ---------------------------------------------------------------------------
// CoordNet (PaiNN-style GNN) for MI455X / gfx1250, wave32 + WMMA f16->f32.
// - Per-edge message MLP hoisted to per-node (row-wise MLP => gather after), a
//   32x GEMM reduction (E=640k edges vs N=20k nodes).
// - All node-state buffers (~110MB) are L2-resident (192MB L2): edge gathers
//   and the 512 fp32 atomics/edge stay on-die.
// - GEMM: block = 8 M-tiles x one 32-col N-tile. Weight tile is staged in LDS
//   pre-packed in WMMA fragment layout (coalesced global loads, b128 LDS
//   reads), A rows load as b128 from global. K=128 fully unrolled: 8
//   v_wmma_f32_16x16x32_f16 per wave.
// ---------------------------------------------------------------------------

typedef __attribute__((ext_vector_type(16))) _Float16 v16h;
typedef __attribute__((ext_vector_type(8)))  float    v8f;

#define HID   128
#define NRBF  20
#define RCUT  5.0f

__device__ __forceinline__ float fast_silu(float x)
{
    // x * sigmoid(x); v_exp_f32 + v_rcp_f32 (no IEEE divide expansion)
    return x * __builtin_amdgcn_rcpf(1.0f + __expf(-x));
}

// ---------------------------------------------------------------------------
// s[n,h] = atom_emb[z[n]-1, h] + charge_emb[clip(charge[batch[n]],-2,2)+2, h]
// ---------------------------------------------------------------------------
__global__ void init_s_kernel(const int* __restrict__ z, const int* __restrict__ charge,
                              const int* __restrict__ batch,
                              const float* __restrict__ atom_emb,
                              const float* __restrict__ charge_emb,
                              float* __restrict__ s, int N)
{
    int idx = blockIdx.x * blockDim.x + threadIdx.x;
    if (idx >= N * HID) return;
    int n = idx >> 7, h = idx & 127;
    int zi = z[n] - 1;
    int c  = charge[batch[n]];
    c = (c < -2 ? -2 : (c > 2 ? 2 : c)) + 2;
    s[idx] = atom_emb[zi * HID + h] + charge_emb[c * HID + h];
}

// ---------------------------------------------------------------------------
// Per-edge geometry: unit vector + distance (computed once, reused 3 layers)
// ---------------------------------------------------------------------------
__global__ void edge_geom_kernel(const int* __restrict__ ei, const int* __restrict__ ej,
                                 const float* __restrict__ pos,
                                 float* __restrict__ ev, float* __restrict__ dist, int E)
{
    int e = blockIdx.x * blockDim.x + threadIdx.x;
    if (e >= E) return;
    int i = ei[e] * 3, j = ej[e] * 3;
    float dx = pos[j+0] - pos[i+0];
    float dy = pos[j+1] - pos[i+1];
    float dz = pos[j+2] - pos[i+2];
    float d  = sqrtf(dx*dx + dy*dy + dz*dz + 1e-12f);
    float inv = 1.0f / d;
    dist[e]   = d;
    ev[e*3+0] = dx * inv;
    ev[e*3+1] = dy * inv;
    ev[e*3+2] = dz * inv;
}

// ---------------------------------------------------------------------------
// WMMA GEMM, K fixed at 128 (K=256 done as two accumulating passes).
//   C[M x Ncols](ldc) = act( [Cin] + A[M x 128](lda) @ W[:, ncol0:+Ncols](wld) [+ bias] )
// grid = (Ncols/32, ceil(ceil(M/16)/8)); block = 256 (8 waves).
// LDS holds the 128x32 weight tile as f16 in fragment order:
//   lbs[frag(2)][kt(4)][lane(32)][q(16)] so a B fragment is one contiguous
//   32-byte read per lane.
// Fragment packing per CDNA5 ISA 7.12.2 (16-bit A 16x32 / B 32x16):
//   lane l: row/col = l&15 ; q<8 -> k = 32*kt + 8*(l>>4)+q ;
//           q>=8    -> k = 32*kt + 16 + 8*(l>>4)+(q-8)
// C layout: lane l, vgpr r -> element (m = r + 8*(l>>4), n = l&15).
// ---------------------------------------------------------------------------
template<int ACT, bool HAS_CIN, bool HAS_BIAS>
__global__ __launch_bounds__(256)
void gemm_f16wmma(const float* __restrict__ A, int lda,
                  const float* __restrict__ Wm, int wld, int ncol0,
                  const float* __restrict__ bias,
                  const float* Cin, int ldcin,       // may alias C
                  float* C, int ldc, int M)
{
    __shared__ _Float16 lbs[2 * 4 * 32 * 16];        // 8 KB
    const int nt    = blockIdx.x;
    const int nbase = nt << 5;
    const int tid   = threadIdx.x;

    // ---- cooperative load of W[:, ncol0+nbase : +32] into fragment layout ----
    #pragma unroll
    for (int it = 0; it < 16; ++it) {
        int e = tid + it * 256;                      // e in [0, 4096)
        int k = e >> 5, c = e & 31;                  // consecutive tid -> consecutive col
        float w = Wm[k * wld + (ncol0 + nbase + c)];
        int kt = k >> 5, kk = k & 31;
        int hs, q;
        if (kk < 16) { hs = kk >> 3; q = kk & 7; }
        else         { hs = (kk - 16) >> 3; q = 8 + ((kk - 16) & 7); }
        int frag = c >> 4, lh = c & 15;
        lbs[(((frag << 2) + kt) << 5 | (lh + (hs << 4))) * 16 + q] = (_Float16)w;
    }
    __syncthreads();

    // ---- per-wave 16x32 output tile ----
    const int wv   = tid >> 5, lane = tid & 31;
    const int ntm  = (M + 15) >> 4;
    const int mt   = blockIdx.y * 8 + wv;
    if (mt >= ntm) return;                           // uniform per-wave: EXEC stays full
    const int mbase = mt << 4;
    const int lh = lane & 15, hs = lane >> 4, kb = hs << 3;

    int arow = mbase + lh; if (arow > M - 1) arow = M - 1;   // clamp; masked at store
    const float* Arow = A + (size_t)(arow * lda);            // 32-bit row offset

    v8f acc0, acc1;
    #pragma unroll
    for (int r = 0; r < 8; ++r) {
        if (HAS_CIN) {
            int m = mbase + r + (hs << 3);
            bool ok = (m < M);
            const float* crow = Cin + (size_t)((ok ? m : 0) * ldcin) + nbase + lh;
            acc0[r] = ok ? crow[0]  : 0.0f;
            acc1[r] = ok ? crow[16] : 0.0f;
        } else {
            acc0[r] = 0.0f; acc1[r] = 0.0f;
        }
    }

    const v16h* bfrag = (const v16h*)lbs;            // 32B-aligned fragments
    #pragma unroll
    for (int kt = 0; kt < 4; ++kt) {
        const float* ap = Arow + kt * 32;
        v16h a;
        #pragma unroll
        for (int q = 0; q < 8; ++q) a[q]     = (_Float16)ap[kb + q];        // contig b128x2
        #pragma unroll
        for (int q = 0; q < 8; ++q) a[8 + q] = (_Float16)ap[16 + kb + q];   // contig b128x2
        v16h b0 = bfrag[(kt << 5) + lane];           // ds_load_b128 x2 (broadcast)
        v16h b1 = bfrag[((4 + kt) << 5) + lane];
        acc0 = __builtin_amdgcn_wmma_f32_16x16x32_f16(false, a, false, b0, (short)0, acc0, false, false);
        acc1 = __builtin_amdgcn_wmma_f32_16x16x32_f16(false, a, false, b1, (short)0, acc1, false, false);
    }

    float bv0 = 0.0f, bv1 = 0.0f;
    if (HAS_BIAS) { bv0 = bias[ncol0 + nbase + lh]; bv1 = bias[ncol0 + nbase + 16 + lh]; }
    #pragma unroll
    for (int r = 0; r < 8; ++r) {
        int m = mbase + r + (hs << 3);
        if (m < M) {
            float v0 = acc0[r] + bv0;
            float v1 = acc1[r] + bv1;
            if (ACT) { v0 = fast_silu(v0); v1 = fast_silu(v1); }
            float* crow = C + (size_t)(m * ldc) + nbase + lh;
            crow[0]  = v0;
            crow[16] = v1;
        }
    }
}

// ---------------------------------------------------------------------------
// Edge message pass. One wave per edge; lane handles h = lane + 32k (k<4).
// W = (rbf @ rbf_proj + b) * cutoff computed in-register (20 MACs per output),
// x = msg[j] * W ; inner = <v[j,:,h], ev> ; scatter fp32 atomics into ds/dv.
// Atomics land on L2-resident ds/dv with lane-contiguous addresses.
// ---------------------------------------------------------------------------
__global__ void edge_msg_kernel(const int* __restrict__ ei, const int* __restrict__ ej,
                                const float* __restrict__ ev, const float* __restrict__ dist,
                                const float* __restrict__ msg, const float* __restrict__ v,
                                const float* __restrict__ rbfW, const float* __restrict__ rbfB,
                                float* ds, float* dv, int E)
{
    int wave = (blockIdx.x * blockDim.x + threadIdx.x) >> 5;
    int lane = threadIdx.x & 31;
    if (wave >= E) return;
    int i = ei[wave], j = ej[wave];
    float d  = dist[wave];
    float e0 = ev[wave*3+0], e1 = ev[wave*3+1], e2 = ev[wave*3+2];

    float cf = (d <= RCUT) ? 0.5f * (1.0f + __cosf(0.62831853071795864769f * d)) : 0.0f;

    const float width    = RCUT / (float)(NRBF - 1);
    const float invwidth = (float)(NRBF - 1) / RCUT;
    float rbf[NRBF];
    #pragma unroll
    for (int r = 0; r < NRBF; ++r) {
        float t = (d - width * (float)r) * invwidth;
        rbf[r] = __expf(-0.5f * t * t);
    }

    const float* mj  = msg + j * 384;                // 32-bit offsets
    const float* vj  = v   + j * 384;
    float*       dsi = ds  + i * HID;
    float*       dvi = dv  + i * 384;
    __builtin_prefetch(mj, 0, 1);                    // -> global_prefetch_b8
    __builtin_prefetch(vj, 0, 1);

    #pragma unroll
    for (int kk = 0; kk < 4; ++kk) {
        int h = lane + 32 * kk;
        float wss = rbfB[h], wsv = rbfB[128 + h], wvv = rbfB[256 + h];
        #pragma unroll
        for (int r = 0; r < NRBF; ++r) {
            float rv = rbf[r];
            const float* wr = rbfW + r * 384;
            wss += rv * wr[h];
            wsv += rv * wr[128 + h];
            wvv += rv * wr[256 + h];
        }
        wss *= cf; wsv *= cf; wvv *= cf;
        float xss = mj[h]       * wss;
        float xsv = mj[128 + h] * wsv;
        float xvv = mj[256 + h] * wvv;
        float inner = e0 * vj[h] + e1 * vj[128 + h] + e2 * vj[256 + h];
        float coef  = xsv + inner * xvv;             // ev ⊗ x_sv + ev ⊗ (inner*x_vv)
        atomicAdd(&dsi[h],        xss);
        atomicAdd(&dvi[h],        e0 * coef);
        atomicAdd(&dvi[128 + h],  e1 * coef);
        atomicAdd(&dvi[256 + h],  e2 * coef);
    }
}

// dst[i] += src[i]
__global__ void vec_add_kernel(float* dst, const float* __restrict__ src, int n)
{
    int i = blockIdx.x * blockDim.x + threadIdx.x;
    if (i < n) dst[i] += src[i];
}

// vvn[n,h] = sqrt( sum_d vvb[n,d,h]^2 + 1e-12 )
__global__ void vvnorm_kernel(const float* __restrict__ vvb, float* __restrict__ vvn, int N)
{
    int idx = blockIdx.x * blockDim.x + threadIdx.x;
    if (idx >= N * HID) return;
    int n = idx >> 7, h = idx & 127;
    const float* p = vvb + n * 384;
    float a = p[h], b = p[128 + h], c = p[256 + h];
    vvn[idx] = sqrtf(a * a + b * b + c * c + 1e-12f);
}

// s += a_ss ; v[n,d] += a_vv*v_u[n,d] + a_sv*v_v[n,d]
__global__ void update_apply_kernel(float* s, float* v,
                                    const float* __restrict__ a,
                                    const float* __restrict__ vu,
                                    const float* __restrict__ vvb, int N)
{
    int idx = blockIdx.x * blockDim.x + threadIdx.x;
    if (idx >= N * HID) return;
    int n = idx >> 7, h = idx & 127;
    const float* an = a + n * 384;
    float ass = an[h], asv = an[128 + h], avv = an[256 + h];
    s[idx] += ass;
    int base = n * 384 + h;
    #pragma unroll
    for (int d = 0; d < 3; ++d) {
        int o = base + d * HID;
        v[o] += avv * vu[o] + asv * vvb[o];
    }
}

// geom[b,h] = mean over the molecule's nodes (batch is sorted, equal counts)
__global__ void pool_kernel(const float* __restrict__ s, float* __restrict__ geom, int npm)
{
    int b = blockIdx.x, h = threadIdx.x;
    const float* p = s + b * npm * HID + h;
    float acc = 0.0f;
    for (int n = 0; n < npm; ++n) acc += p[n * HID];
    geom[b * HID + h] = acc * (1.0f / (float)npm);
}

// out[m] = bias[0] + sum_k X[m,0:128]*w[k]   (one wave per row)
__global__ void rowdot_kernel(const float* __restrict__ X, const float* __restrict__ w,
                              const float* __restrict__ bptr, float* __restrict__ out, int M)
{
    int row  = (blockIdx.x * blockDim.x + threadIdx.x) >> 5;
    int lane = threadIdx.x & 31;
    if (row >= M) return;
    const float* x = X + row * HID;
    float acc = x[lane] * w[lane] + x[lane + 32] * w[lane + 32]
              + x[lane + 64] * w[lane + 64] + x[lane + 96] * w[lane + 96];
    #pragma unroll
    for (int off = 16; off > 0; off >>= 1) acc += __shfl_xor(acc, off, 32);
    if (lane == 0) out[row] = acc + bptr[0];
}

// dipole[b] = || sum_{n in b} (v[n,:,:] . dipole_w) ||
__global__ void dipole_kernel(const float* __restrict__ v, const float* __restrict__ dw,
                              float* __restrict__ out, int npm)
{
    __shared__ float red[3][HID];
    int b = blockIdx.x, t = threadIdx.x;
    const float* base = v + b * npm * 384;
    float w = dw[t];
    float a0 = 0.f, a1 = 0.f, a2 = 0.f;
    for (int n = 0; n < npm; ++n) {
        const float* vn = base + n * 384;
        a0 += vn[t] * w; a1 += vn[128 + t] * w; a2 += vn[256 + t] * w;
    }
    red[0][t] = a0; red[1][t] = a1; red[2][t] = a2;
    __syncthreads();
    for (int st = 64; st > 0; st >>= 1) {
        if (t < st) {
            red[0][t] += red[0][t + st];
            red[1][t] += red[1][t + st];
            red[2][t] += red[2][t + st];
        }
        __syncthreads();
    }
    if (t == 0) {
        float x = red[0][0], y = red[1][0], z = red[2][0];
        out[b] = sqrtf(x * x + y * y + z * z + 1e-12f);
    }
}

// ---------------------------------------------------------------------------
// Host side. Param leaves follow setup_inputs() dict insertion order,
// recursively (tuples/lists in order): 57 leaves total.
// ---------------------------------------------------------------------------
extern "C" void kernel_launch(void* const* d_in, const int* in_sizes, int n_in,
                              void* d_out, int out_size, void* d_ws, size_t ws_size,
                              hipStream_t stream)
{
    (void)n_in; (void)out_size; (void)ws_size;
    const int N   = in_sizes[0];
    const int B   = in_sizes[2];
    const int E   = in_sizes[4] / 2;
    const int npm = N / B;

    const int*   z      = (const int*)d_in[0];
    const float* pos    = (const float*)d_in[1];
    const int*   charge = (const int*)d_in[2];
    const int*   batch  = (const int*)d_in[3];
    const int*   ei     = (const int*)d_in[4];
    const int*   ej     = ei + E;
    const float* atom_emb   = (const float*)d_in[5];
    const float* charge_emb = (const float*)d_in[6];

    const float *msg1w[3], *msg1b[3], *msg2w[3], *msg2b[3], *rbfw[3], *rbfb[3],
                *vecw[3], *upd1w[3], *upd1b[3], *upd2w[3], *upd2b[3];
    for (int l = 0; l < 3; ++l) {
        int o = 7 + 11 * l;
        msg1w[l] = (const float*)d_in[o + 0];  msg1b[l] = (const float*)d_in[o + 1];
        msg2w[l] = (const float*)d_in[o + 2];  msg2b[l] = (const float*)d_in[o + 3];
        rbfw[l]  = (const float*)d_in[o + 4];  rbfb[l]  = (const float*)d_in[o + 5];
        vecw[l]  = (const float*)d_in[o + 6];
        upd1w[l] = (const float*)d_in[o + 7];  upd1b[l] = (const float*)d_in[o + 8];
        upd2w[l] = (const float*)d_in[o + 9];  upd2b[l] = (const float*)d_in[o + 10];
    }
    const float* r1w = (const float*)d_in[40]; const float* r1b = (const float*)d_in[41];
    const float* r2w = (const float*)d_in[42]; const float* r2b = (const float*)d_in[43];
    const float* e1w = (const float*)d_in[44]; const float* e1b = (const float*)d_in[45];
    const float* e2w = (const float*)d_in[46]; const float* e2b = (const float*)d_in[47];
    const float* g1w = (const float*)d_in[48]; const float* g1b = (const float*)d_in[49];
    const float* g2w = (const float*)d_in[50]; const float* g2b = (const float*)d_in[51];
    const float* nn1w= (const float*)d_in[52]; const float* nn1b= (const float*)d_in[53];
    const float* nn2w= (const float*)d_in[54]; const float* nn2b= (const float*)d_in[55];
    const float* dipw= (const float*)d_in[56];

    // Workspace layout (floats); ~205 MB total.
    float* Wf = (float*)d_ws;
    size_t o = 0;
    float* s    = Wf + o; o += (size_t)N * HID;
    float* v    = Wf + o; o += (size_t)N * 3 * HID;
    float* ds   = Wf + o; o += (size_t)N * HID;
    float* dv   = Wf + o; o += (size_t)N * 3 * HID;
    float* h1   = Wf + o; o += (size_t)N * HID;        // MLP hidden (reused)
    float* big  = Wf + o; o += (size_t)N * 3 * HID;    // msg / 'a' output (reused)
    float* vu   = Wf + o; o += (size_t)N * 3 * HID;
    float* vvb  = Wf + o; o += (size_t)N * 3 * HID;
    float* vvn  = Wf + o; o += (size_t)N * HID;
    float* evv  = Wf + o; o += (size_t)E * 3;
    float* dst  = Wf + o; o += (size_t)E;
    float* geom = Wf + o; o += (size_t)B * HID;
    float* hg   = Wf + o; o += (size_t)B * HID;
    float* tg   = Wf + o; o += (size_t)B * HID;

    float* out = (float*)d_out;   // [energy(B) | gap(B) | dipole(B) | npa(N)]

    // dispatch: templated GEMM (K=128 fixed). Combos used:
    //   act=1 bias, act=0 bias, act=0 no-bias, act=1 bias+Cin
    auto gemm = [&](const float* A, int lda, const float* Wm, int wld, int ncol0,
                    const float* bias, const float* Cin, int ldcin,
                    float* C, int ldc, int M, int Ncols, int act) {
        int ntm = (M + 15) / 16;
        dim3 g(Ncols / 32, (ntm + 7) / 8), bl(256);
        if (Cin) {
            if (act) hipLaunchKernelGGL((gemm_f16wmma<1, true, true>), g, bl, 0, stream,
                                        A, lda, Wm, wld, ncol0, bias, Cin, ldcin, C, ldc, M);
            else     hipLaunchKernelGGL((gemm_f16wmma<0, true, true>), g, bl, 0, stream,
                                        A, lda, Wm, wld, ncol0, bias, Cin, ldcin, C, ldc, M);
        } else if (bias) {
            if (act) hipLaunchKernelGGL((gemm_f16wmma<1, false, true>), g, bl, 0, stream,
                                        A, lda, Wm, wld, ncol0, bias, nullptr, 0, C, ldc, M);
            else     hipLaunchKernelGGL((gemm_f16wmma<0, false, true>), g, bl, 0, stream,
                                        A, lda, Wm, wld, ncol0, bias, nullptr, 0, C, ldc, M);
        } else {
            hipLaunchKernelGGL((gemm_f16wmma<0, false, false>), g, bl, 0, stream,
                               A, lda, Wm, wld, ncol0, nullptr, nullptr, 0, C, ldc, M);
        }
    };
    auto add = [&](float* dstp, const float* srcp, int n) {
        hipLaunchKernelGGL(vec_add_kernel, dim3((n + 255) / 256), dim3(256), 0, stream,
                           dstp, srcp, n);
    };

    // ---- init ----
    hipMemsetAsync(v, 0, (size_t)N * 3 * HID * sizeof(float), stream);
    hipLaunchKernelGGL(init_s_kernel, dim3((N * HID + 255) / 256), dim3(256), 0, stream,
                       z, charge, batch, atom_emb, charge_emb, s, N);
    hipLaunchKernelGGL(edge_geom_kernel, dim3((E + 255) / 256), dim3(256), 0, stream,
                       ei, ej, pos, evv, dst, E);

    // ---- message-passing layers ----
    for (int l = 0; l < 3; ++l) {
        // node-side message MLP (hoisted out of the edge loop): msg = mlp(s)
        gemm(s,  HID, msg1w[l], HID, 0, msg1b[l], nullptr, 0, h1,  HID, N, HID, 1);
        gemm(h1, HID, msg2w[l], 384, 0, msg2b[l], nullptr, 0, big, 384, N, 384, 0);

        hipMemsetAsync(ds, 0, (size_t)N * HID * sizeof(float), stream);
        hipMemsetAsync(dv, 0, (size_t)N * 3 * HID * sizeof(float), stream);
        hipLaunchKernelGGL(edge_msg_kernel, dim3((E + 7) / 8), dim3(256), 0, stream,
                           ei, ej, evv, dst, big, v, rbfw[l], rbfb[l], ds, dv, E);
        add(s, ds, N * HID);
        add(v, dv, N * 3 * HID);

        // update: vproj split into v_u / v_v (v viewed as [3N,128] row-major)
        gemm(v, HID, vecw[l], 256, 0,   nullptr, nullptr, 0, vu,  HID, 3 * N, HID, 0);
        gemm(v, HID, vecw[l], 256, 128, nullptr, nullptr, 0, vvb, HID, 3 * N, HID, 0);
        hipLaunchKernelGGL(vvnorm_kernel, dim3((N * HID + 255) / 256), dim3(256), 0, stream,
                           vvb, vvn, N);
        // concat(s, vvn) @ upd1 as two accumulating K=128 passes (no concat buffer)
        gemm(s,   HID, upd1w[l],             HID, 0, nullptr,  nullptr, 0,  h1, HID, N, HID, 0);
        gemm(vvn, HID, upd1w[l] + HID * HID, HID, 0, upd1b[l], h1, HID,     h1, HID, N, HID, 1);
        gemm(h1,  HID, upd2w[l], 384, 0, upd2b[l], nullptr, 0, big, 384, N, 384, 0);
        hipLaunchKernelGGL(update_apply_kernel, dim3((N * HID + 255) / 256), dim3(256), 0, stream,
                           s, v, big, vu, vvb, N);
    }

    // ---- readout ----
    hipLaunchKernelGGL(pool_kernel, dim3(B), dim3(HID), 0, stream, s, geom, npm);
    gemm(geom, HID, r1w, HID, 0, r1b, nullptr, 0, hg, HID, B, HID, 1);
    gemm(hg,   HID, r2w, HID, 0, r2b, nullptr, 0, tg, HID, B, HID, 0);   // h

    gemm(tg, HID, e1w, HID, 0, e1b, nullptr, 0, hg, HID, B, HID, 1);
    hipLaunchKernelGGL(rowdot_kernel, dim3((B + 7) / 8), dim3(256), 0, stream,
                       hg, e2w, e2b, out, B);                                  // energy
    gemm(tg, HID, g1w, HID, 0, g1b, nullptr, 0, hg, HID, B, HID, 1);
    hipLaunchKernelGGL(rowdot_kernel, dim3((B + 7) / 8), dim3(256), 0, stream,
                       hg, g2w, g2b, out + B, B);                              // gap

    hipLaunchKernelGGL(dipole_kernel, dim3(B), dim3(HID), 0, stream,
                       v, dipw, out + 2 * B, npm);                             // dipole

    gemm(s, HID, nn1w, HID, 0, nn1b, nullptr, 0, h1, HID, N, HID, 1);
    hipLaunchKernelGGL(rowdot_kernel, dim3((N + 7) / 8), dim3(256), 0, stream,
                       h1, nn2w, nn2b, out + 3 * B, N);                        // npa
}
